// QuantLinearBase_79388175499416
// MI455X (gfx1250) — 8-bit matrix operand order repaired and checked by reading, compiled, not run on hardware
//
#include <hip/hip_runtime.h>
#include <hip/hip_bf16.h>


typedef int v8i __attribute__((ext_vector_type(8)));

#define TILE_M 128
#define TILE_N 128
#define TILE_K 64
#define LDA 80
#define LDB 144

__device__ __forceinline__ void async_copy_b128(unsigned lds_off, const void* gsrc) {
    unsigned long long ga = (unsigned long long)(uintptr_t)gsrc;
    asm volatile("global_load_async_to_lds_b128 %0, %1, off"
                 :: "v"(lds_off), "v"(ga) : "memory");
}
__device__ __forceinline__ void wait_async_le4() { asm volatile("s_wait_asynccnt 0x4" ::: "memory"); }
__device__ __forceinline__ void wait_async_0()  { asm volatile("s_wait_asynccnt 0x0" ::: "memory"); }

__global__ void init_ws_kernel(unsigned int* maxbits) {
    *maxbits = 0u;
}

__global__ __launch_bounds__(256)
void maxabs_kernel(const float* __restrict__ x, unsigned int* __restrict__ maxbits,
                   long n4) {
    __shared__ float red[256];
    const int tid = threadIdx.x;
    float m = 0.0f;
    const float4* x4 = (const float4*)x;
    for (long i = (long)blockIdx.x * blockDim.x + tid; i < n4;
         i += (long)gridDim.x * blockDim.x) {
        float4 v = x4[i];
        m = fmaxf(m, fmaxf(fmaxf(fabsf(v.x), fabsf(v.y)),
                           fmaxf(fabsf(v.z), fabsf(v.w))));
    }
    red[tid] = m;
    __syncthreads();
    for (int s = 128; s > 0; s >>= 1) {
        if (tid < s) red[tid] = fmaxf(red[tid], red[tid + s]);
        __syncthreads();
    }
    if (tid == 0) {
        atomicMax(maxbits, __float_as_uint(red[0]));
    }
}

__global__ __launch_bounds__(256)
void quant_kernel(const float* __restrict__ x, signed char* __restrict__ xq,
                  const unsigned int* __restrict__ maxbits, long nthreads) {
    long id = (long)blockIdx.x * blockDim.x + threadIdx.x;
    if (id >= nthreads) return;
    const float maxabs = __uint_as_float(*maxbits);
    const float scale = (maxabs == 0.0f) ? 1.0f : maxabs * (1.0f / 127.0f);
    const float inv = 1.0f / scale;

    const float4* src = (const float4*)(x + id * 16);
    int packed[4];
#pragma unroll
    for (int g = 0; g < 4; ++g) {
        float4 v = src[g];
        int q0 = __float2int_rn(v.x * inv);
        int q1 = __float2int_rn(v.y * inv);
        int q2 = __float2int_rn(v.z * inv);
        int q3 = __float2int_rn(v.w * inv);
        q0 = max(-127, min(127, q0)); q1 = max(-127, min(127, q1));
        q2 = max(-127, min(127, q2)); q3 = max(-127, min(127, q3));
        packed[g] = (q0 & 0xff) | ((q1 & 0xff) << 8) | ((q2 & 0xff) << 16) | (q3 << 24);
    }
    *(int4*)(xq + id * 16) = make_int4(packed[0], packed[1], packed[2], packed[3]);
}

__global__ __launch_bounds__(256)
void unpack_kernel(const int* __restrict__ qw, signed char* __restrict__ wt,
                   int O, int I2) {
    long id = (long)blockIdx.x * blockDim.x + threadIdx.x;
    int o = (int)(id % O);
    int j = (int)(id / O);
    int q = qw[(long)o * I2 + j] & 0xff;
    signed char hi = (signed char)(((q >> 4) & 15) - 8);
    signed char lo = (signed char)((q & 15) - 8);
    wt[(long)(2 * j)     * O + o] = hi;
    wt[(long)(2 * j + 1) * O + o] = lo;
}

__global__ __launch_bounds__(256)
void gemm_iu8_kernel(const signed char* __restrict__ xq,
                     const signed char* __restrict__ wt,
                     const float* __restrict__ bias,
                     const float* __restrict__ wscale,
                     const unsigned int* __restrict__ maxbits,
                     float* __restrict__ out,
                     int M, int N, int K) {
    __shared__ __align__(16) signed char smA[2][TILE_M * LDA];
    __shared__ __align__(16) signed char smB[2][TILE_K * LDB];

    const int tid  = threadIdx.x;
    const int lane = tid & 31;
    const int wid  = tid >> 5;
    const int wm   = (wid >> 2) * 64;
    const int wn   = (wid & 3)  * 32;
    const int m0   = blockIdx.y * TILE_M;
    const int n0   = blockIdx.x * TILE_N;
    const int l15  = lane & 15;
    const int half8 = (lane >> 4) << 3;
    const int aoff  = (lane >> 4) << 3;

    v8i acc[4][2];
#pragma unroll
    for (int i = 0; i < 4; ++i)
#pragma unroll
        for (int j = 0; j < 2; ++j)
            acc[i][j] = (v8i){0, 0, 0, 0, 0, 0, 0, 0};

    const int rowA = tid >> 2, colA = (tid & 3) << 4;
    const int rowB = tid >> 3, colB = (tid & 7) << 4;
    const signed char* gA0 = xq + (long)(m0 + rowA) * K + colA;
    const signed char* gA1 = gA0 + (long)64 * K;
    const signed char* gB0 = wt + (long)rowB * N + n0 + colB;
    const signed char* gB1 = gB0 + (long)32 * N;

    unsigned lA[2], lB[2];
#pragma unroll
    for (int b = 0; b < 2; ++b) {
        lA[b] = (unsigned)(uintptr_t)&smA[b][rowA * LDA + colA];
        lB[b] = (unsigned)(uintptr_t)&smB[b][rowB * LDB + colB];
    }

    auto stage = [&](int buf) {
        async_copy_b128(lA[buf],                 gA0);
        async_copy_b128(lA[buf] + 64u * LDA,     gA1);
        async_copy_b128(lB[buf],                 gB0);
        async_copy_b128(lB[buf] + 32u * LDB,     gB1);
        gA0 += TILE_K;            gA1 += TILE_K;
        gB0 += (long)TILE_K * N;  gB1 += (long)TILE_K * N;
    };

    auto compute = [&](int buf) {
        v8i bfrag[2];
#pragma unroll
        for (int nsub = 0; nsub < 2; ++nsub) {
            int noff = wn + nsub * 16;
            union { v8i v; signed char c[32]; } ub;
#pragma unroll
            for (int s = 0; s < 32; ++s)
                ub.c[s] = smB[buf][((s >> 3) * 16 + aoff + (s & 7)) * LDB + noff + l15];
            bfrag[nsub] = ub.v;
        }
        v8i afrag[4];
#pragma unroll
        for (int msub = 0; msub < 4; ++msub) {
            int base = (wm + msub * 16 + l15) * LDA + aoff;
            union { v8i v; long long d[4]; } ua;
            ua.d[0] = *(const long long*)(smA[buf] + base);
            ua.d[1] = *(const long long*)(smA[buf] + base + 16);
            ua.d[2] = *(const long long*)(smA[buf] + base + 32);
            ua.d[3] = *(const long long*)(smA[buf] + base + 48);
            afrag[msub] = ua.v;
        }
#pragma unroll
        for (int msub = 0; msub < 4; ++msub)
#pragma unroll
            for (int nsub = 0; nsub < 2; ++nsub)
                acc[msub][nsub] = __builtin_amdgcn_wmma_i32_16x16x64_iu8(
                     true, afrag[msub],  true, bfrag[nsub],
                    acc[msub][nsub],  false,  false);
    };

    stage(0);

    const int KT = K / TILE_K;
    for (int it = 0; it < KT - 1; ++it) {
        const int buf = it & 1;
        stage(buf ^ 1);
        wait_async_le4();
        __syncthreads();
        compute(buf);
        __syncthreads();
    }
    wait_async_0();
    __syncthreads();
    compute((KT - 1) & 1);

    const float maxabs = __uint_as_float(*maxbits);
    const float ascale = (maxabs == 0.0f) ? 1.0f : maxabs * (1.0f / 127.0f);
    const float total  = ascale * wscale[0];

#pragma unroll
    for (int msub = 0; msub < 4; ++msub) {
#pragma unroll
        for (int nsub = 0; nsub < 2; ++nsub) {
            int col = n0 + wn + nsub * 16 + l15;
            float b = bias[col];
#pragma unroll
            for (int r = 0; r < 8; ++r) {
                int row = m0 + wm + msub * 16 + r + half8;
                out[(long)row * N + col] = (float)acc[msub][nsub][r] * total + b;
            }
        }
    }
}

extern "C" void kernel_launch(void* const* d_in, const int* in_sizes, int n_in,
                              void* d_out, int out_size, void* d_ws, size_t ws_size,
                              hipStream_t stream) {
    const float* x       = (const float*)d_in[0];
    const int*   qweight = (const int*)d_in[1];
    const float* wscale  = (const float*)d_in[2];
    const float* bias    = (const float*)d_in[3];
    float* out = (float*)d_out;

    const int K  = 4096;
    const int N  = 4096;
    const int M  = in_sizes[0] / K;
    const int I2 = K / 2;

    unsigned int* maxbits = (unsigned int*)d_ws;
    signed char*  xq = (signed char*)d_ws + 256;
    signed char*  wt = (signed char*)d_ws + 256 + (size_t)M * K;

    init_ws_kernel<<<1, 1, 0, stream>>>(maxbits);

    long n4 = (long)M * K / 4;
    maxabs_kernel<<<2048, 256, 0, stream>>>(x, maxbits, n4);

    long wthreads = (long)N * I2;
    unpack_kernel<<<(int)(wthreads / 256), 256, 0, stream>>>(qweight, wt, N, I2);

    long qthreads = (long)M * K / 16;
    quant_kernel<<<(int)(qthreads / 256), 256, 0, stream>>>(x, xq, maxbits, qthreads);

    dim3 grid(N / TILE_N, M / TILE_M);
    gemm_iu8_kernel<<<grid, 256, 0, stream>>>(xq, wt, bias, wscale, maxbits,
                                              out, M, N, K);
}
